// CrossAttentionProjectLayer_78065325572368
// MI455X (gfx1250) — compile-verified, matrix operand order checked by reading, not run on hardware
//
#include <hip/hip_runtime.h>
#include <hip/hip_bf16.h>

// ---------------- problem constants ----------------
#define S_TOT  4096
#define B_TOT  8
#define E_TOT  1024
#define H_TOT  16
#define HD     64     // head_dim
#define PD     64     // proj_dim
#define PD2    128    // 2*proj_dim
#define TILE_S 128    // rows per workgroup
#define EC     128    // staged K-chunk (elements) for GEMM1 B operand

typedef unsigned short u16;
typedef __attribute__((ext_vector_type(16))) __bf16 v16bf;
typedef __attribute__((ext_vector_type(8)))  float  v8f;
typedef int v4i_vs __attribute__((vector_size(16)));   // matches builtin param

#if defined(__has_builtin)
#if __has_builtin(__builtin_amdgcn_global_load_async_to_lds_b128) && \
    __has_builtin(__builtin_amdgcn_s_wait_asynccnt)
#define HAVE_ASYNC_LDS 1
#endif
#endif

__device__ __forceinline__ u16 f2bf(float f) {
  unsigned u = __float_as_uint(f);
  u += 0x7FFFu + ((u >> 16) & 1u);          // round-to-nearest-even
  return (u16)(u >> 16);
}
__device__ __forceinline__ float bf2f(u16 v) {
  return __uint_as_float(((unsigned)v) << 16);
}

union ABPack { uint4 q[2]; v16bf v; };

// Load one 16-bit WMMA A/B operand fragment: two contiguous 16-byte chunks,
// second chunk 16 elements (32B) after the first (CDNA5 16-bit A/B layout:
// lanes 0-15 hold K=c0..c0+7 and c0+16..c0+23 with c0=0, lanes 16-31 with c0=8).
__device__ __forceinline__ v16bf ld_frag(const u16* p) {
  ABPack u;
  u.q[0] = *(const uint4*)(p);
  u.q[1] = *(const uint4*)(p + 16);
  return u.v;
}

__device__ __forceinline__ v8f wmma_bf16(v16bf a, v16bf b, v8f c) {
  return __builtin_amdgcn_wmma_f32_16x16x32_bf16(
      /*neg_a=*/false, a, /*neg_b=*/false, b,
      /*c_mod=*/(short)0, c, /*reuse_a=*/false, /*reuse_b=*/false);
}

// one 16-byte global -> LDS transfer (async on CDNA5, sync fallback otherwise)
__device__ __forceinline__ void copy16_g2l(const u16* g, u16* l) {
#if defined(HAVE_ASYNC_LDS)
  __builtin_amdgcn_global_load_async_to_lds_b128(
      (__attribute__((address_space(1))) v4i_vs*)(g),
      (__attribute__((address_space(3))) v4i_vs*)(l),
      /*offset=*/0, /*cpol=*/0);
#else
  *(uint4*)l = *(const uint4*)g;
#endif
}

// Cooperative stage of a 64-row x EC-col bf16 weight chunk into LDS.
// 64*EC/8 = 1024 16B segments; 256 threads x 4 segments, coalesced.
__device__ __forceinline__ void stage_wchunk(const u16* __restrict__ wbase,
                                             int h64, int e0, u16* buf, int tid) {
#pragma unroll
  for (int q = 0; q < 4; ++q) {
    const int seg = q * 256 + tid;
    const int n   = seg >> 4;          // row 0..63
    const int c8  = (seg & 15) * 8;    // col 0..120
    copy16_g2l(wbase + (size_t)(h64 + n) * E_TOT + e0 + c8, buf + n * EC + c8);
  }
}

__device__ __forceinline__ void wait_stage_pending() {
#if defined(HAVE_ASYNC_LDS)
  __builtin_amdgcn_s_wait_asynccnt(4);   // current chunk (4 ops/lane) complete
#endif
}
__device__ __forceinline__ void wait_stage_all() {
#if defined(HAVE_ASYNC_LDS)
  __builtin_amdgcn_s_wait_asynccnt(0);
#endif
}

// ---------------- prep: weights -> bf16, rm = sigma*R/tau ----------------
__global__ __launch_bounds__(256)
void prep_kernel(const float* __restrict__ kw, const float* __restrict__ vw,
                 const float* __restrict__ rmat, const float* __restrict__ sigma,
                 u16* __restrict__ kwb, u16* __restrict__ vwb,
                 u16* __restrict__ rmb, float* __restrict__ out_rm) {
  const int i = blockIdx.x * 256 + threadIdx.x;       // grid covers E*E exactly
  kwb[i] = f2bf(kw[i]);
  vwb[i] = f2bf(vw[i]);
  if (i < H_TOT * PD * HD) {
    const int h = i >> 12;            // / (PD*HD)
    const int d = i & (HD - 1);
    const float rv = sigma[h * HD + d] * rmat[i];     // TAU == 1.0
    out_rm[i] = rv;
    rmb[i] = f2bf(rv);
  }
}

// ---------------- encoder f32 -> bf16 ----------------
__global__ __launch_bounds__(256)
void convert_enc_kernel(const float* __restrict__ in, u16* __restrict__ out, int n8) {
  const int i = blockIdx.x * 256 + threadIdx.x;
  if (i >= n8) return;
  const float4* p = (const float4*)in;
  const float4 a = p[2 * (size_t)i];
  const float4 b = p[2 * (size_t)i + 1];
  union { u16 s[8]; uint4 q; } u;
  u.s[0] = f2bf(a.x); u.s[1] = f2bf(a.y); u.s[2] = f2bf(a.z); u.s[3] = f2bf(a.w);
  u.s[4] = f2bf(b.x); u.s[5] = f2bf(b.y); u.s[6] = f2bf(b.z); u.s[7] = f2bf(b.w);
  *(uint4*)(out + (size_t)i * 8) = u.q;
}

// ---------------- fused main ----------------
// grid: (S/TILE_S, B, H), block: 256 (8 waves)
__global__ __launch_bounds__(256)
void fused_main(const u16* __restrict__ enc_bf,
                const u16* __restrict__ kw_bf,
                const u16* __restrict__ vw_bf,
                const u16* __restrict__ rm_bf,
                const float* __restrict__ kb,
                const float* __restrict__ vb,
                const unsigned char* __restrict__ mask,
                float* __restrict__ out_s,
                float* __restrict__ out_z) {
  __shared__ u16 lds_pk [TILE_S * HD];   // pk[m][d]                     16 KB
  __shared__ u16 lds_pvT[HD * TILE_S];   // pv^T[d][m]                   16 KB
  __shared__ u16 lds_phi[PD2 * TILE_S];  // phi^T[k2][m] / weight stage  32 KB

  const int tid  = threadIdx.x;
  const int lane = tid & 31;
  const int wv   = tid >> 5;        // wave 0..7
  const int lrow = lane & 15;       // M (A) or N (B) within tile
  const int ghi  = lane >> 4;       // lane half
  const int c0   = ghi * 8;         // K chunk base per lane half
  const int s0   = blockIdx.x * TILE_S;
  const int b    = blockIdx.y;
  const int h    = blockIdx.z;
  const int h64  = h * 64;

  // per-lane mask flags for the 8 rows this lane's accumulator covers
  unsigned char mrow[8];
#pragma unroll
  for (int i = 0; i < 8; ++i) {
    const int m = wv * 16 + ghi * 8 + i;
    mrow[i] = mask[(size_t)(s0 + m) * B_TOT + b];
  }

  const u16* encRow =
      enc_bf + ((size_t)(s0 + wv * 16 + lrow) * B_TOT + b) * E_TOT;

  const v8f vzero = {0.f, 0.f, 0.f, 0.f, 0.f, 0.f, 0.f, 0.f};

  // ---------- GEMM1: pk / pv = enc @ W^T + bias (K = 1024) ----------
  // B operand (weights) is shared by all 8 waves: double-buffer-stage E-chunks
  // of the 64 weight rows into LDS (async global->LDS on CDNA5).
  u16* const wstage0 = lds_phi;                  // 64*EC u16 = 16 KB
  u16* const wstage1 = lds_phi + 64 * EC;        // 64*EC u16 = 16 KB

  for (int kv = 0; kv < 2; ++kv) {
    const u16*   wbase = kv ? vw_bf : kw_bf;
    const float* bias  = kv ? vb : kb;

    v8f acc[4];
#pragma unroll
    for (int nt = 0; nt < 4; ++nt) acc[nt] = vzero;

    stage_wchunk(wbase, h64, 0, wstage0, tid);

    const int NSTEP = E_TOT / EC;               // 8
    for (int st = 0; st < NSTEP; ++st) {
      u16* const buf = (st & 1) ? wstage1 : wstage0;
      if (st + 1 < NSTEP) {
        stage_wchunk(wbase, h64, (st + 1) * EC, (st & 1) ? wstage0 : wstage1, tid);
        wait_stage_pending();                   // chunk st landed (in-order)
      } else {
        wait_stage_all();
      }
      __syncthreads();                          // chunk st visible to all waves

      const u16* aBase = encRow + st * EC;
#pragma unroll
      for (int ke = 0; ke < EC; ke += 32) {
        const v16bf a = ld_frag(aBase + ke + c0);
#pragma unroll
        for (int nt = 0; nt < 4; ++nt) {
          const v16bf bm = ld_frag(buf + (nt * 16 + lrow) * EC + ke + c0);
          acc[nt] = wmma_bf16(a, bm, acc[nt]);
        }
      }
      __syncthreads();                          // done reading buf before reuse
    }

#pragma unroll
    for (int nt = 0; nt < 4; ++nt) {
      const float bvv = bias[h64 + nt * 16 + lrow];
      const int   d   = nt * 16 + lrow;
#pragma unroll
      for (int i = 0; i < 8; ++i) {
        const int m = wv * 16 + ghi * 8 + i;
        const u16 bs = f2bf(acc[nt][i] + bvv);
        if (kv == 0) lds_pk[m * HD + d] = bs;
        else         lds_pvT[d * TILE_S + m] = bs;
      }
    }
  }
  __syncthreads();

  // ---------- GEMM2: proj = pk @ rm_h^T (K = 64), then phi ----------
  {
    v8f acc[4];
#pragma unroll
    for (int nt = 0; nt < 4; ++nt) acc[nt] = vzero;

    const u16* aRow = lds_pk + (wv * 16 + lrow) * HD;
#pragma unroll
    for (int d0 = 0; d0 < HD; d0 += 32) {
      const v16bf a = ld_frag(aRow + d0 + c0);
#pragma unroll
      for (int nt = 0; nt < 4; ++nt) {
        const u16* r = rm_bf + ((size_t)h * PD + nt * 16 + lrow) * HD;
        const v16bf bm = ld_frag(r + d0 + c0);
        acc[nt] = wmma_bf16(a, bm, acc[nt]);
      }
    }
    const float scale = 0.125f;   // PROJ_DIM^-0.5
    __syncthreads();              // weight stage fully consumed; reuse as phi
#pragma unroll
    for (int nt = 0; nt < 4; ++nt) {
      const int k = nt * 16 + lrow;
#pragma unroll
      for (int i = 0; i < 8; ++i) {
        const int m = wv * 16 + ghi * 8 + i;
        float sv, cv;
        __sincosf(acc[nt][i], &sv, &cv);
        sv *= scale; cv *= scale;
        if (mrow[i]) { sv = 0.f; cv = 0.f; }
        lds_phi[k * TILE_S + m]        = f2bf(sv);
        lds_phi[(k + PD) * TILE_S + m] = f2bf(cv);
      }
    }
  }
  __syncthreads();

  // ---------- z partials (wave-uniform branch: waves 0-3) ----------
  if (tid < PD2) {
    float zs = 0.f;
#pragma unroll 8
    for (int m = 0; m < TILE_S; ++m) zs += bf2f(lds_phi[tid * TILE_S + m]);
    unsafeAtomicAdd(out_z + ((size_t)b * H_TOT + h) * PD2 + tid, zs);
  }

  // ---------- GEMM3: s_bh += phi^T @ pv (K = 128) ----------
  {
    v8f acc[4];
#pragma unroll
    for (int nt = 0; nt < 4; ++nt) acc[nt] = vzero;

    const u16* phiRow = lds_phi + (wv * 16 + lrow) * TILE_S;
#pragma unroll
    for (int m0 = 0; m0 < TILE_S; m0 += 32) {
      const v16bf a = ld_frag(phiRow + m0 + c0);
#pragma unroll
      for (int nt = 0; nt < 4; ++nt) {
        const v16bf bm = ld_frag(lds_pvT + (nt * 16 + lrow) * TILE_S + m0 + c0);
        acc[nt] = wmma_bf16(a, bm, acc[nt]);
      }
    }
    float* sb = out_s + ((size_t)b * H_TOT + h) * PD2 * HD;
#pragma unroll
    for (int nt = 0; nt < 4; ++nt) {
      const int d = nt * 16 + lrow;
#pragma unroll
      for (int i = 0; i < 8; ++i) {
        const int k2 = wv * 16 + ghi * 8 + i;
        unsafeAtomicAdd(sb + (size_t)k2 * HD + d, acc[nt][i]);
      }
    }
  }
}

// ---------------- host launcher ----------------
extern "C" void kernel_launch(void* const* d_in, const int* in_sizes, int n_in,
                              void* d_out, int out_size, void* d_ws, size_t ws_size,
                              hipStream_t stream) {
  const float* enc   = (const float*)d_in[0];
  const float* rmat  = (const float*)d_in[1];
  const float* kw    = (const float*)d_in[2];
  const float* kb    = (const float*)d_in[3];
  const float* vw    = (const float*)d_in[4];
  const float* vb    = (const float*)d_in[5];
  const float* sigma = (const float*)d_in[6];
  const unsigned char* mask = (const unsigned char*)d_in[7];

  float* out    = (float*)d_out;
  float* out_s  = out;                                         // [B,H,128,64]
  float* out_z  = out + (size_t)B_TOT * H_TOT * PD2 * HD;      // [B,H,128]
  float* out_rm = out_z + (size_t)B_TOT * H_TOT * PD2;         // [H,64,64]

  u16* ws   = (u16*)d_ws;
  u16* kwb  = ws;                                              // 1M elems
  u16* vwb  = ws + (size_t)E_TOT * E_TOT;                      // 1M elems
  u16* rmb  = ws + (size_t)2 * E_TOT * E_TOT;                  // 64K elems
  u16* encb = rmb + (size_t)H_TOT * PD * HD;                   // 32M elems

  // zero the atomically-accumulated outputs (s and z)
  (void)hipMemsetAsync(d_out, 0,
                       (size_t)(B_TOT * H_TOT * PD2 * HD + B_TOT * H_TOT * PD2) *
                           sizeof(float),
                       stream);

  prep_kernel<<<dim3((E_TOT * E_TOT) / 256), 256, 0, stream>>>(
      kw, vw, rmat, sigma, kwb, vwb, rmb, out_rm);

  const int n8 = (S_TOT * B_TOT * E_TOT) / 8;
  convert_enc_kernel<<<dim3(n8 / 256), 256, 0, stream>>>(enc, encb, n8);

  fused_main<<<dim3(S_TOT / TILE_S, B_TOT, H_TOT), 256, 0, stream>>>(
      encb, kwb, vwb, rmb, kb, vb, mask, out_s, out_z);
}
